// Conv2dModulated_45071386804770
// MI455X (gfx1250) — matrix operand order repaired and checked by reading, not compiled, not run on hardware
//
#include <hip/hip_runtime.h>
#include <hip/hip_bf16.h>

typedef __attribute__((ext_vector_type(16))) _Float16 v16h;
typedef __attribute__((ext_vector_type(8)))  _Float16 v8h;
typedef __attribute__((ext_vector_type(8)))  float    v8f;

#define B_    4
#define CIN   64
#define COUT  64
#define Hn    512
#define Wn    512
#define Ln    512
#define KTOT  576      // 9 taps * 64 cin, K index = tap*64 + cin
#define WPITCH 584     // padded K pitch (f16): 1168B -> 36-word lane stride, conflict-free
#define TW    32
#define TH    4
#define NT    4        // y-tiles per workgroup (reuse LDS weights)
#define PW    (TW + 2) // 34 (halo)
#define PH    (TH + 2) // 6
#define CPITCH 72      // per-pixel f16 pitch (64 cin + 8 pad): 144B -> conflict-free

// ---------------------------------------------------------------- styles
__global__ void k_styles(const float* __restrict__ w, const float* __restrict__ aw,
                         const float* __restrict__ ab, float* __restrict__ s) {
    int t = threadIdx.x;          // 256 threads: (b, cin)
    int b = t >> 6, ci = t & 63;
    const float* wr = w + b * Ln;
    const float* ar = aw + ci * Ln;
    float acc = 0.f;
    #pragma unroll 4
    for (int l = 0; l < Ln; ++l) acc += wr[l] * ar[l];
    s[t] = acc + ab[ci] + 1.0f;
}

// ---------------- modulate + demodulate, written directly in LDS layout
__global__ void k_modw(const float* __restrict__ weight, const float* __restrict__ s,
                       _Float16* __restrict__ wm) {
    __shared__ float red[64];
    int co = blockIdx.x & 63, b = blockIdx.x >> 6;
    int ci = threadIdx.x;                       // 64 threads
    float sc = s[b * 64 + ci];
    const float* wp = weight + (co * CIN + ci) * 9;
    float wv[9], sq = 0.f;
    #pragma unroll
    for (int k = 0; k < 9; ++k) { float v = sc * wp[k]; wv[k] = v; sq += v * v; }
    red[ci] = sq;
    __syncthreads();
    if (ci == 0) {
        float tot = 0.f;
        #pragma unroll
        for (int i = 0; i < 64; ++i) tot += red[i];
        red[0] = rsqrtf(tot + 1e-8f);
    }
    __syncthreads();
    float d = red[0];
    // padded pitch: global image == LDS image, enables linear async block copy
    _Float16* dst = wm + (size_t)(b * COUT + co) * WPITCH + ci;   // K = tap*64 + ci
    #pragma unroll
    for (int k = 0; k < 9; ++k) dst[k * 64] = (_Float16)(d * wv[k]);
}

// ---------------------------------------------------- implicit-GEMM conv
__global__ void __launch_bounds__(256)
k_conv(const float* __restrict__ x, const _Float16* __restrict__ wm,
       float* __restrict__ out) {
    extern __shared__ __align__(16) char smem[];
    _Float16* xl = (_Float16*)smem;                              // PH*PW*CPITCH f16
    _Float16* wl = (_Float16*)(smem + PH * PW * CPITCH * 2);     // COUT*WPITCH f16

    const int tid = threadIdx.x;
    const int b   = blockIdx.z;
    const int x0  = blockIdx.x * TW;

    // stage 64x584 f16 weight block via async global->LDS DMA (no VGPR bounce)
    {
        const char* src  = (const char*)(wm + (size_t)b * COUT * WPITCH);
        uint32_t    dstl = (uint32_t)(uintptr_t)wl;              // LDS byte address
        const int   NCHUNK = COUT * WPITCH * 2 / 16;             // 4672 x b128
        for (int i = tid; i < NCHUNK; i += 256) {
            uint64_t g = (uint64_t)(uintptr_t)(src + (size_t)i * 16);
            uint32_t l = dstl + i * 16;
            asm volatile("global_load_async_to_lds_b128 %0, %1, off"
                         :: "v"(l), "v"(g) : "memory");
        }
        asm volatile("s_wait_asynccnt 0x0" ::: "memory");
    }

    const int lane = tid & 31, wv = tid >> 5;
    const int m0   = (wv & 3) * 16;      // Cout block of this wave
    const int pg   = wv >> 2;            // pixel-group (2 rows of 32 each)
    const int ncol = lane & 15, hk = lane >> 4;
    const float* xb = x + (size_t)b * CIN * Hn * Wn;

    for (int t = 0; t < NT; ++t) {
        const int y0 = (blockIdx.y * NT + t) * TH;
        __syncthreads();
        // stage x tile with halo: 2 cin planes -> packed f16x2 b32 LDS store
        const int PLANE = PH * PW;                               // 204
        for (int i = tid; i < (CIN / 2) * PLANE; i += 256) {
            int c2 = i / PLANE, r = i % PLANE;
            int py = r / PW, px = r % PW;
            int gy = y0 - 1 + py, gx = x0 - 1 + px;
            float v0 = 0.f, v1 = 0.f;
            if ((unsigned)gy < Hn && (unsigned)gx < Wn) {
                const float* gp = xb + (((size_t)(c2 * 2)) * Hn + gy) * Wn + gx;
                v0 = gp[0];
                v1 = gp[(size_t)Hn * Wn];
                if (t + 1 < NT && gy + TH < Hn) {
                    __builtin_prefetch(gp + TH * Wn, 0, 1);                    // global_prefetch_b8
                    __builtin_prefetch(gp + (size_t)Hn * Wn + TH * Wn, 0, 1);
                }
            }
            union { _Float16 h[2]; uint32_t u; } pk;
            pk.h[0] = (_Float16)v0; pk.h[1] = (_Float16)v1;
            *(uint32_t*)(xl + (py * PW + px) * CPITCH + c2 * 2) = pk.u;
        }
        __syncthreads();

        v8f acc[4] = {};
        #pragma unroll
        for (int kk = 0; kk < 18; ++kk) {            // K chunks of 32
            const int tap = kk >> 1, hc = kk & 1;
            const int dy = tap / 3, dx = tap % 3;
            // A fragment: lane m=ncol; VGPR0-3 hold K=hk*8.., VGPR4-7 hold K=16+hk*8..
            const _Float16* ap = wl + (m0 + ncol) * WPITCH + kk * 32 + hk * 8;
            v8h alo = *(const v8h*)ap;
            v8h ahi = *(const v8h*)(ap + 16);
            v16h afrag = __builtin_shufflevector(alo, ahi,
                0, 1, 2, 3, 4, 5, 6, 7, 8, 9, 10, 11, 12, 13, 14, 15);
            const int koff = hc * 32 + hk * 8;       // B: lane holds K=hk*8..+7 and 16+hk*8..+7
            #pragma unroll
            for (int j = 0; j < 4; ++j) {
                const int py = pg * 2 + (j >> 1) + dy;
                const int px = (j & 1) * 16 + ncol + dx;
                const _Float16* bp = xl + (py * PW + px) * CPITCH + koff;
                v8h blo = *(const v8h*)bp;
                v8h bhi = *(const v8h*)(bp + 16);
                v16h bfrag = __builtin_shufflevector(blo, bhi,
                    0, 1, 2, 3, 4, 5, 6, 7, 8, 9, 10, 11, 12, 13, 14, 15);
                acc[j] = __builtin_amdgcn_wmma_f32_16x16x32_f16(
                    false, afrag, false, bfrag, (short)0, acc[j], false, false);
            }
        }

        // write out: C layout -> VGPR g: lanes0-15 M=g, lanes16-31 M=8+g
        #pragma unroll
        for (int j = 0; j < 4; ++j) {
            const int gy = y0 + pg * 2 + (j >> 1);
            const int gx = x0 + (j & 1) * 16 + ncol;
            #pragma unroll
            for (int g = 0; g < 8; ++g) {
                int co = m0 + hk * 8 + g;
                out[(((size_t)b * COUT + co) * Hn + gy) * Wn + gx] = acc[j][g];
            }
        }
        __syncthreads();
    }
}

// ---------------------------------------------------------------- launch
extern "C" void kernel_launch(void* const* d_in, const int* in_sizes, int n_in,
                              void* d_out, int out_size, void* d_ws, size_t ws_size,
                              hipStream_t stream) {
    (void)in_sizes; (void)n_in; (void)out_size; (void)ws_size;
    const float* x      = (const float*)d_in[0];
    const float* w      = (const float*)d_in[1];
    const float* weight = (const float*)d_in[2];
    const float* aw     = (const float*)d_in[3];
    const float* ab     = (const float*)d_in[4];
    float* out = (float*)d_out;

    float*     styles = (float*)d_ws;                         // 4*64 f32 = 1KB
    _Float16*  wmod   = (_Float16*)((char*)d_ws + 1024);      // 4*64*584 f16 (padded pitch)

    k_styles<<<1, 256, 0, stream>>>(w, aw, ab, styles);
    k_modw<<<B_ * COUT, 64, 0, stream>>>(weight, styles, wmod);

    size_t shmem = (size_t)(PH * PW * CPITCH + COUT * WPITCH) * sizeof(_Float16);
    dim3 grid(Wn / TW, Hn / (TH * NT), B_);
    k_conv<<<grid, 256, shmem, stream>>>(x, wmod, out);
}
